// Informer_68942815035993
// MI455X (gfx1250) — compile-verified, matrix operand order checked by reading, not run on hardware
//
#include <hip/hip_runtime.h>
#include <cmath>

// ---------------------------------------------------------------------------
// Informer forward for MI455X (gfx1250, wave32, WMMA).
// GEMMs: 4-wave blocks, 32x64 output tile. A k-slices (32x64) staged in LDS
// through a DOUBLE-BUFFERED GLOBAL_LOAD_ASYNC_TO_LDS_B128 pipeline
// (issue tile i+1, s_wait_asynccnt 1, compute tile i) so the async DMA
// overlaps the 4 register-blocked v_wmma_f32_16x16x32_f16 per k-step.
// Irregular ProbSparse bookkeeping -> small VALU/LDS kernels.
// ---------------------------------------------------------------------------

#define D_MODEL 512
#define HD      64
#define NH      8
#define BSZ     4
#define LENC    2048
#define LDEC    2
#define KSTEP   64

typedef __attribute__((ext_vector_type(16))) _Float16 v16h;
typedef __attribute__((ext_vector_type(8)))  float    v8f;

// ======================= WMMA GEMM =========================================
// C[M,N] = A[M,K] * W[K,N] (+bias[n]) (+res[m,n]) (gelu?) -> f32 and/or f16.
// A: f16, row-major, lda = Kpad (mult of 64, rows 16B aligned).
// WT: f16, N-major (WT[n*ldw + k]), ldw = Kpad, zero padded.
// Block: 128 threads (4 waves). Block tile: 32(M) x 64(N).
__global__ __launch_bounds__(128) void gemm16(
    const _Float16* __restrict__ A, int lda,
    const _Float16* __restrict__ WT, int ldw,
    const float* __restrict__ bias,
    const float* __restrict__ res, int ldr,
    float* __restrict__ Cf, _Float16* __restrict__ Ch, int ldc,
    int M, int N, int K, int gelu)
{
    // two LDS buffers of one 32x64-half A slice each (4 KB per buffer)
    __shared__ __align__(16) _Float16 sA[2][32 * KSTEP];

    int tid  = threadIdx.x;
    int wave = tid >> 5;
    int lane = tid & 31;
    int hlf  = lane >> 4;
    int idx  = lane & 15;

    int tm = blockIdx.y * 32;
    int tn = blockIdx.x * 64 + wave * 16;

    // loader mapping: 256 16-byte chunks per tile, 2 per thread
    int r0 = tid >> 3,        c0 = tid & 7;          // rows 0..15
    int r1 = (tid + 128) >> 3, c1 = tid & 7;         // rows 16..31
    int g0 = tm + r0; if (g0 >= M) g0 = M - 1;       // clamp (store-guarded)
    int g1 = tm + r1; if (g1 >= M) g1 = M - 1;
    const _Float16* As0 = A + (size_t)g0 * lda + c0 * 8;
    const _Float16* As1 = A + (size_t)g1 * lda + c1 * 8;
    unsigned lo0[2], lo1[2];
    lo0[0] = (unsigned)(size_t)&sA[0][r0 * KSTEP + c0 * 8];
    lo0[1] = (unsigned)(size_t)&sA[1][r0 * KSTEP + c0 * 8];
    lo1[0] = (unsigned)(size_t)&sA[0][r1 * KSTEP + c1 * 8];
    lo1[1] = (unsigned)(size_t)&sA[1][r1 * KSTEP + c1 * 8];

    int colW = tn + idx; if (colW >= N) colW = N - 1;
    const _Float16* Wp = WT + (size_t)colW * ldw + hlf * 8;

    v8f acc[2];
    #pragma unroll
    for (int s = 0; s < 2; s++)
        #pragma unroll
        for (int v = 0; v < 8; v++) acc[s][v] = 0.f;

    union U16 { v16h v; uint4 q[2]; };

    auto issue = [&](int buf, int k0) {
        unsigned long long ga0 = (unsigned long long)(size_t)(As0 + k0);
        unsigned long long ga1 = (unsigned long long)(size_t)(As1 + k0);
        asm volatile("global_load_async_to_lds_b128 %0, %1, off"
                     :: "v"(lo0[buf]), "v"(ga0) : "memory");
        asm volatile("global_load_async_to_lds_b128 %0, %1, off"
                     :: "v"(lo1[buf]), "v"(ga1) : "memory");
    };

    int nt = K / KSTEP;
    issue(0, 0);                                     // prologue

    for (int i = 0; i < nt; i++) {
        int k0 = i * KSTEP;
        int cur = i & 1;
        if (i + 1 < nt) {
            issue(cur ^ 1, k0 + KSTEP);              // overlap next DMA
            asm volatile("s_wait_asynccnt 1" ::: "memory");   // tile i landed
        } else {
            asm volatile("s_wait_asynccnt 0" ::: "memory");
        }
        __builtin_prefetch(Wp + k0 + 2 * KSTEP, 0, 1);
        __syncthreads();

        const _Float16* sa = &sA[cur][0];
        #pragma unroll
        for (int s2 = 0; s2 < 2; s2++) {
            U16 b;
            b.q[0] = *(const uint4*)(Wp + k0 + s2 * 32);
            b.q[1] = *(const uint4*)(Wp + k0 + s2 * 32 + 16);
            U16 a0, a1;
            a0.q[0] = *(const uint4*)&sa[idx * KSTEP + s2 * 32 + hlf * 8];
            a0.q[1] = *(const uint4*)&sa[idx * KSTEP + s2 * 32 + 16 + hlf * 8];
            a1.q[0] = *(const uint4*)&sa[(16 + idx) * KSTEP + s2 * 32 + hlf * 8];
            a1.q[1] = *(const uint4*)&sa[(16 + idx) * KSTEP + s2 * 32 + 16 + hlf * 8];
            acc[0] = __builtin_amdgcn_wmma_f32_16x16x32_f16(
                         false, a0.v, false, b.v, (short)0, acc[0], false, false);
            acc[1] = __builtin_amdgcn_wmma_f32_16x16x32_f16(
                         false, a1.v, false, b.v, (short)0, acc[1], false, false);
        }
        __syncthreads();                             // readers done before overwrite
    }

    int col = tn + idx;
    if (col < N) {
        float bv = bias ? bias[col] : 0.f;
        #pragma unroll
        for (int s = 0; s < 2; s++) {
            #pragma unroll
            for (int v = 0; v < 8; v++) {
                int m = tm + s * 16 + hlf * 8 + v;
                if (m < M) {
                    float x = acc[s][v] + bv;
                    if (res)  x += res[(size_t)m * ldr + col];
                    if (gelu) x = 0.5f * x * (1.0f + erff(x * 0.70710678118f));
                    if (Cf) Cf[(size_t)m * ldc + col] = x;
                    if (Ch) Ch[(size_t)m * ldc + col] = (_Float16)x;
                }
            }
        }
    }
}

// W[K,N] f32 -> WT[Npad, Kpad] f16 (zero padded)
__global__ void wt_build(const float* __restrict__ W, _Float16* __restrict__ WT,
                         int K, int N, int Kpad, int Npad)
{
    size_t i = (size_t)blockIdx.x * blockDim.x + threadIdx.x;
    size_t total = (size_t)Npad * Kpad;
    if (i >= total) return;
    int k = (int)(i % Kpad);
    int n = (int)(i / Kpad);
    float v = (k < K && n < N) ? W[(size_t)k * N + n] : 0.f;
    WT[i] = (_Float16)v;
}

__global__ void cvt_half(const float* __restrict__ src, _Float16* __restrict__ dst, size_t n)
{
    size_t i = (size_t)blockIdx.x * blockDim.x + threadIdx.x;
    if (i < n) dst[i] = (_Float16)src[i];
}

// circular-conv gather: Aaug[b*L+t][kk*C+c] = src[b, (t-1+kk)%L, c], pad to Kpad
__global__ void aug_build(const float* __restrict__ src, _Float16* __restrict__ dst,
                          int Bn, int L, int C, int Kpad)
{
    size_t i = (size_t)blockIdx.x * blockDim.x + threadIdx.x;
    size_t total = (size_t)Bn * L * Kpad;
    if (i >= total) return;
    int k = (int)(i % Kpad);
    size_t row = i / Kpad;
    int t = (int)(row % L);
    int b = (int)(row / L);
    float v = 0.f;
    if (k < 3 * C) {
        int kk = k / C, c = k % C;
        int ts = (t - 1 + kk + L) % L;
        v = src[((size_t)(b * L + ts)) * C + c];
    }
    dst[i] = (_Float16)v;
}

// X = conv_val + pos_emb + (x_mark @ tw + tb); also f16 mirror
__global__ void emb_finish(const float* __restrict__ val, const float* __restrict__ xmark,
                           const float* __restrict__ tw, const float* __restrict__ tb,
                           float* __restrict__ X, _Float16* __restrict__ Xh, int Bn, int L)
{
    size_t i = (size_t)blockIdx.x * blockDim.x + threadIdx.x;
    size_t total = (size_t)Bn * L * D_MODEL;
    if (i >= total) return;
    int d = (int)(i % D_MODEL);
    size_t row = i / D_MODEL;
    int t = (int)(row % L);
    int i2 = d >> 1;
    float ang = (float)t * expf(-(float)(2 * i2) * 0.0179889460f); // ln(1e4)/512
    float pe = (d & 1) ? cosf(ang) : sinf(ang);
    float tem = tb[d];
    for (int f = 0; f < 6; f++) tem += xmark[row * 6 + f] * tw[f * D_MODEL + d];
    float y = val[i] + pe + tem;
    X[i] = y;
    Xh[i] = (_Float16)y;
}

// decoder embedding (L=2, DEC_IN=1): conv + pos + temporal, all in one
__global__ void dec_emb_k(const float* __restrict__ xdec, const float* __restrict__ cw,
                          const float* __restrict__ xmark, const float* __restrict__ tw,
                          const float* __restrict__ tb,
                          float* __restrict__ X, _Float16* __restrict__ Xh, int Bn, int L)
{
    int i = blockIdx.x * blockDim.x + threadIdx.x;
    int total = Bn * L * D_MODEL;
    if (i >= total) return;
    int d = i % D_MODEL;
    int row = i / D_MODEL;
    int t = row % L;
    int b = row / L;
    float acc = 0.f;
    for (int kk = 0; kk < 3; kk++) {
        int ts = (t - 1 + kk + L) % L;
        acc += xdec[b * L + ts] * cw[kk * D_MODEL + d];
    }
    int i2 = d >> 1;
    float ang = (float)t * expf(-(float)(2 * i2) * 0.0179889460f);
    float pe = (d & 1) ? cosf(ang) : sinf(ang);
    float tem = tb[d];
    for (int f = 0; f < 6; f++) tem += xmark[row * 6 + f] * tw[f * D_MODEL + d];
    float y = acc + pe + tem;
    X[i] = y;
    Xh[i] = (_Float16)y;
}

// LayerNorm over D=512, one 256-thread block per row; writes f32 + f16
__global__ void layernorm_k(const float* __restrict__ X, const float* __restrict__ g,
                            const float* __restrict__ b, float* __restrict__ Y,
                            _Float16* __restrict__ Yh, int rows)
{
    __shared__ float s1[256], s2[256];
    int row = blockIdx.x;
    int tid = threadIdx.x;
    const float* xr = X + (size_t)row * D_MODEL;
    float a0 = xr[tid], a1 = xr[tid + 256];
    s1[tid] = a0 + a1;
    s2[tid] = a0 * a0 + a1 * a1;
    __syncthreads();
    for (int s = 128; s > 0; s >>= 1) {
        if (tid < s) { s1[tid] += s1[tid + s]; s2[tid] += s2[tid + s]; }
        __syncthreads();
    }
    float mu  = s1[0] * (1.f / 512.f);
    float var = s2[0] * (1.f / 512.f) - mu * mu;
    float inv = rsqrtf(var + 1e-5f);
    for (int d = tid; d < D_MODEL; d += 256) {
        float y = (xr[d] - mu) * inv * g[d] + b[d];
        Y[(size_t)row * D_MODEL + d] = y;
        if (Yh) Yh[(size_t)row * D_MODEL + d] = (_Float16)y;
    }
}

// mean of V over L per (b, h*64+d)
__global__ void vmean_k(const float* __restrict__ V, float* __restrict__ vm, int Bn, int L)
{
    int i = blockIdx.x * blockDim.x + threadIdx.x;
    if (i >= Bn * D_MODEL) return;
    int b = i / D_MODEL, c = i % D_MODEL;
    float s = 0.f;
    for (int l = 0; l < L; l++) s += V[((size_t)(b * L + l)) * D_MODEL + c];
    vm[i] = s / (float)L;
}

__global__ void ctx_init_k(const float* __restrict__ vm, float* __restrict__ ctx, int Bn, int L)
{
    size_t i = (size_t)blockIdx.x * blockDim.x + threadIdx.x;
    size_t total = (size_t)Bn * L * D_MODEL;
    if (i >= total) return;
    int c = (int)(i % D_MODEL);
    int b = (int)((i / D_MODEL) / L);
    ctx[i] = vm[b * D_MODEL + c];
}

// sparsity measure M = max_s(QK_s) - mean over L_K (deterministic LCG sampling)
__global__ void sample_m(const float* __restrict__ Q, const float* __restrict__ Kk,
                         float* __restrict__ Mb, int Bn, int L, int LK, int ns)
{
    int tid = blockIdx.x * blockDim.x + threadIdx.x;
    if (tid >= Bn * NH * L) return;
    int lq = tid % L;
    int bh = tid / L;
    int b = bh >> 3, h = bh & 7;
    const float* qp = Q + ((size_t)(b * L + lq) * D_MODEL + h * HD);
    float mx = -3.4e38f, sm = 0.f;
    for (int s = 0; s < ns; s++) {
        unsigned ks = (1103515245u * (unsigned)lq + 2654435761u * (unsigned)s + 12345u)
                      % (unsigned)LK;
        const float* kp = Kk + ((size_t)(b * LK + (int)ks) * D_MODEL + h * HD);
        float acc = 0.f;
        #pragma unroll 8
        for (int d = 0; d < HD; d++) acc += qp[d] * kp[d];
        mx = fmaxf(mx, acc);
        sm += acc;
    }
    Mb[(size_t)bh * L + lq] = mx - sm / (float)LK;
}

// per-(b,h) top-u of M via iterative LDS argmax
__global__ void topk_k(const float* __restrict__ Mb, int* __restrict__ Mtop, int L, int u)
{
    __shared__ float sm[2048];
    __shared__ float rv[256];
    __shared__ int   ri[256];
    int bh = blockIdx.x, tid = threadIdx.x;
    const float* src = Mb + (size_t)bh * L;
    for (int k = tid; k < L; k += 256) sm[k] = src[k];
    __syncthreads();
    for (int it = 0; it < u; it++) {
        float bv = -3.4e38f; int bi = 0;
        for (int k = tid; k < L; k += 256)
            if (sm[k] > bv) { bv = sm[k]; bi = k; }
        rv[tid] = bv; ri[tid] = bi;
        __syncthreads();
        for (int s = 128; s > 0; s >>= 1) {
            if (tid < s && rv[tid + s] > rv[tid]) { rv[tid] = rv[tid + s]; ri[tid] = ri[tid + s]; }
            __syncthreads();
        }
        if (tid == 0) { Mtop[bh * u + it] = ri[0]; sm[ri[0]] = -3.4e38f; }
        __syncthreads();
    }
}

// full softmax-attention update for selected queries (Mtop==null -> q=ui)
__global__ void attn_update(const float* __restrict__ Q, const float* __restrict__ Kk,
                            const float* __restrict__ V, const int* __restrict__ Mtop,
                            float* __restrict__ ctx, int L, int LK, int u)
{
    __shared__ float sc[2048];
    __shared__ float red[256];
    int blk = blockIdx.x;
    int ui = blk % u;
    int bh = blk / u;
    int b = bh >> 3, h = bh & 7;
    int tid = threadIdx.x;
    int q = Mtop ? Mtop[bh * u + ui] : ui;
    const float* qp = Q + ((size_t)(b * L + q) * D_MODEL + h * HD);

    for (int k = tid; k < LK; k += 256) {
        const float* kp = Kk + ((size_t)(b * LK + k) * D_MODEL + h * HD);
        float acc = 0.f;
        #pragma unroll 8
        for (int d = 0; d < HD; d++) acc += qp[d] * kp[d];
        sc[k] = acc * 0.125f;                         // 1/sqrt(64)
    }
    __syncthreads();
    float lm = -3.4e38f;
    for (int k = tid; k < LK; k += 256) lm = fmaxf(lm, sc[k]);
    red[tid] = lm; __syncthreads();
    for (int s = 128; s > 0; s >>= 1) {
        if (tid < s) red[tid] = fmaxf(red[tid], red[tid + s]);
        __syncthreads();
    }
    float mx = red[0];
    __syncthreads();
    float ls = 0.f;
    for (int k = tid; k < LK; k += 256) { float e = expf(sc[k] - mx); sc[k] = e; ls += e; }
    red[tid] = ls; __syncthreads();
    for (int s = 128; s > 0; s >>= 1) {
        if (tid < s) red[tid] += red[tid + s];
        __syncthreads();
    }
    float ssum = red[0];
    __syncthreads();
    int d = tid & 63, grp = tid >> 6;
    float acc = 0.f;
    for (int k = grp; k < LK; k += 4)
        acc += sc[k] * V[((size_t)(b * LK + k) * D_MODEL + h * HD + d)];
    red[tid] = acc; __syncthreads();
    if (tid < 64) {
        float o = (red[tid] + red[tid + 64] + red[tid + 128] + red[tid + 192]) / ssum;
        ctx[((size_t)(b * L + q) * D_MODEL + h * HD + tid)] = o;
    }
}

// decoder causal self-attention, L=2: row0 = V0, row1 = softmax([q1k0,q1k1]/8) @ V
__global__ void dec_self_attn(const float* __restrict__ Q, const float* __restrict__ Kk,
                              const float* __restrict__ V, float* __restrict__ ctx, int Bn)
{
    int i = blockIdx.x * blockDim.x + threadIdx.x;
    if (i >= Bn * NH * HD) return;
    int d = i % HD;
    int h = (i / HD) & 7;
    int b = i / (NH * HD);
    const float* q1 = Q  + ((size_t)(b * 2 + 1) * D_MODEL + h * HD);
    const float* k0 = Kk + ((size_t)(b * 2 + 0) * D_MODEL + h * HD);
    const float* k1 = Kk + ((size_t)(b * 2 + 1) * D_MODEL + h * HD);
    float s0 = 0.f, s1 = 0.f;
    for (int e = 0; e < HD; e++) { s0 += q1[e] * k0[e]; s1 += q1[e] * k1[e]; }
    s0 *= 0.125f; s1 *= 0.125f;
    float mx = fmaxf(s0, s1);
    float p0 = expf(s0 - mx), p1 = expf(s1 - mx);
    float inv = 1.f / (p0 + p1);
    float v0 = V[((size_t)(b * 2 + 0) * D_MODEL + h * HD + d)];
    float v1 = V[((size_t)(b * 2 + 1) * D_MODEL + h * HD + d)];
    ctx[((size_t)(b * 2 + 0) * D_MODEL + h * HD + d)] = v0;
    ctx[((size_t)(b * 2 + 1) * D_MODEL + h * HD + d)] = (p0 * v0 + p1 * v1) * inv;
}

// batchnorm affine + ELU + maxpool(k=3, s=2, pad=1) over L
__global__ void bn_elu_pool(const float* __restrict__ Y, const float* __restrict__ g,
                            const float* __restrict__ beta, const float* __restrict__ rm,
                            const float* __restrict__ rv, float* __restrict__ Xo,
                            _Float16* __restrict__ Xoh, int Bn, int L, int Lo)
{
    size_t i = (size_t)blockIdx.x * blockDim.x + threadIdx.x;
    size_t total = (size_t)Bn * Lo * D_MODEL;
    if (i >= total) return;
    int d = (int)(i % D_MODEL);
    size_t row = i / D_MODEL;
    int to = (int)(row % Lo);
    int b = (int)(row / Lo);
    float sc = rsqrtf(rv[d] + 1e-5f) * g[d];
    float mu = rm[d], bb = beta[d];
    float m = -3.4e38f;
    for (int j = -1; j <= 1; j++) {
        int t = 2 * to + j;
        if (t < 0 || t >= L) continue;
        float v = (Y[((size_t)(b * L + t)) * D_MODEL + d] - mu) * sc + bb;
        float e = v > 0.f ? v : (expf(v) - 1.f);
        m = fmaxf(m, e);
    }
    Xo[i] = m;
    Xoh[i] = (_Float16)m;
}

__global__ void out_copy(const float* __restrict__ P, float* __restrict__ out, int Bn)
{
    int i = blockIdx.x * blockDim.x + threadIdx.x;
    if (i >= Bn * 118) return;
    int b = i / 118, c = i % 118;
    out[i] = P[(size_t)(b * LDEC + (LDEC - 1)) * 118 + c];
}

// ======================= host orchestration ================================
extern "C" void kernel_launch(void* const* d_in, const int* in_sizes, int n_in,
                              void* d_out, int out_size, void* d_ws, size_t ws_size,
                              hipStream_t stream)
{
    (void)in_sizes; (void)n_in; (void)out_size; (void)ws_size;
    const size_t MB = 1u << 20;
    char* ws = (char*)d_ws;

    // --- workspace layout (~168 MB) ---
    float*    X    = (float*)(ws + 0);          // 16 MB   activations f32 (max 8192x512)
    _Float16* Xh   = (_Float16*)(ws + 16 * MB); //  8 MB   activations f16
    float*    R    = (float*)(ws + 24 * MB);    // 16 MB   gemm temp
    float*    Qb   = (float*)(ws + 40 * MB);    // 16 MB
    float*    Kb   = (float*)(ws + 56 * MB);    // 16 MB
    float*    Vb   = (float*)(ws + 72 * MB);    // 16 MB
    float*    CTX  = (float*)(ws + 88 * MB);    // 16 MB
    _Float16* CTXh = (_Float16*)(ws + 104 * MB);//  8 MB
    _Float16* Hh   = (_Float16*)(ws + 112 * MB);// 32 MB   FFN hidden f16 / conv-augment
    _Float16* WTb  = (_Float16*)(ws + 144 * MB);//  4 MB   transposed f16 weights
    float*    Mb   = (float*)(ws + 148 * MB);   // 256 KB  sparsity measure
    int*      Mtop = (int*)(ws + 149 * MB);     //  <4 KB
    float*    VM   = (float*)(ws + 150 * MB);   //  8 KB
    float*    Kc   = (float*)(ws + 151 * MB);   //  8 MB   cross-attn K
    float*    Vc   = (float*)(ws + 159 * MB);   //  8 MB   cross-attn V
    char* dp = ws + 167 * MB;                   // decoder small buffers
    float*    dX    = (float*)dp;              dp += 8 * 512 * 4;
    _Float16* dXh   = (_Float16*)dp;           dp += 8 * 512 * 2;
    float*    dR    = (float*)dp;              dp += 8 * 512 * 4;
    float*    dQ    = (float*)dp;              dp += 8 * 512 * 4;
    float*    dK    = (float*)dp;              dp += 8 * 512 * 4;
    float*    dV    = (float*)dp;              dp += 8 * 512 * 4;
    float*    dCTX  = (float*)dp;              dp += 8 * 512 * 4;
    _Float16* dCTXh = (_Float16*)dp;           dp += 8 * 512 * 2;
    _Float16* dHh   = (_Float16*)dp;           dp += 8 * 2048 * 2;
    float*    Pbuf  = (float*)dp;              dp += 8 * 128 * 4;

    auto in = [&](int i) { return (const float*)d_in[i]; };
    // input index map (recursive insertion-order flattening of setup_inputs()):
    // 0 x_enc  1 x_mark_enc  2 x_dec  3 x_mark_dec
    // 4 enc_emb.conv_w  5 tw  6 tb    7 dec_emb.conv_w  8 tw  9 tb
    // 10..17 enc0.attn(wq,bq,wk,bk,wv,bv,wo,bo)
    // 18 w1 19 b1 20 w2 21 b2 22 ln1_g 23 ln1_b 24 ln2_g 25 ln2_b 26 ln3_g 27 ln3_b
    // 28..45 enc1 (same pattern)  46..51 distil0(w,b,g,beta,rm,rv)
    // 52 enc_norm_g 53 enc_norm_b
    // 54..61 dec0.sattn  62..69 dec0.cattn
    // 70 w1 71 b1 72 w2 73 b2 74 ln1_g 75 ln1_b 76 ln2_g 77 ln2_b 78 ln3_g 79 ln3_b
    // 80 dec_norm_g 81 dec_norm_b  82 proj_w 83 proj_b

    auto gemm = [&](const _Float16* A, int lda, const float* W, int K, int N,
                    int Kpad, int Npad, const float* bias, const float* res, int ldr,
                    float* Cf, _Float16* Ch, int ldc, int M, int gelu) {
        size_t wtot = (size_t)Npad * Kpad;
        wt_build<<<dim3((unsigned)((wtot + 255) / 256)), dim3(256), 0, stream>>>(
            W, WTb, K, N, Kpad, Npad);
        dim3 grid((N + 63) / 64, (M + 31) / 32);
        gemm16<<<grid, dim3(128), 0, stream>>>(A, lda, WTb, Kpad, bias, res, ldr,
                                               Cf, Ch, ldc, M, N, Kpad, gelu);
    };

    auto ew = [&](size_t n) { return dim3((unsigned)((n + 255) / 256)); };

    // ---------------- encoder embedding ----------------
    {
        int L = LENC, N = BSZ * L;
        aug_build<<<ew((size_t)N * 384), 256, 0, stream>>>(in(0), Hh, BSZ, L, 118, 384);
        gemm(Hh, 384, in(4), 354, 512, 384, 512, nullptr, nullptr, 0,
             R, nullptr, 512, N, 0);
        emb_finish<<<ew((size_t)N * 512), 256, 0, stream>>>(R, in(1), in(5), in(6),
                                                            X, Xh, BSZ, L);
    }

    // ---------------- encoder layer -------------------
    auto enc_layer = [&](int L, int ab, int fb) {
        int N = BSZ * L;
        int uv = 3 * (int)ceil(log((double)L)); if (uv > L) uv = L;
        // QKV projections
        gemm(Xh, 512, in(ab + 0), 512, 512, 512, 512, in(ab + 1), nullptr, 0, Qb, nullptr, 512, N, 0);
        gemm(Xh, 512, in(ab + 2), 512, 512, 512, 512, in(ab + 3), nullptr, 0, Kb, nullptr, 512, N, 0);
        gemm(Xh, 512, in(ab + 4), 512, 512, 512, 512, in(ab + 5), nullptr, 0, Vb, nullptr, 512, N, 0);
        // ProbSparse attention
        vmean_k<<<ew(BSZ * 512), 256, 0, stream>>>(Vb, VM, BSZ, L);
        ctx_init_k<<<ew((size_t)N * 512), 256, 0, stream>>>(VM, CTX, BSZ, L);
        sample_m<<<ew((size_t)BSZ * NH * L), 256, 0, stream>>>(Qb, Kb, Mb, BSZ, L, L, uv);
        topk_k<<<BSZ * NH, 256, 0, stream>>>(Mb, Mtop, L, uv);
        attn_update<<<BSZ * NH * uv, 256, 0, stream>>>(Qb, Kb, Vb, Mtop, CTX, L, L, uv);
        cvt_half<<<ew((size_t)N * 512), 256, 0, stream>>>(CTX, CTXh, (size_t)N * 512);
        // out projection + residual, ln1
        gemm(CTXh, 512, in(ab + 6), 512, 512, 512, 512, in(ab + 7), X, 512, R, nullptr, 512, N, 0);
        layernorm_k<<<N, 256, 0, stream>>>(R, in(fb + 4), in(fb + 5), X, Xh, N);
        // FFN (gelu in epilogue, hidden kept f16 only) + residual, ln2
        gemm(Xh, 512, in(fb + 0), 512, 2048, 512, 2048, in(fb + 1), nullptr, 0,
             nullptr, Hh, 2048, N, 1);
        gemm(Hh, 2048, in(fb + 2), 2048, 512, 2048, 512, in(fb + 3), X, 512,
             R, nullptr, 512, N, 0);
        layernorm_k<<<N, 256, 0, stream>>>(R, in(fb + 6), in(fb + 7), X, Xh, N);
    };

    enc_layer(LENC, 10, 18);

    // ---------------- distil conv layer ----------------
    {
        int L = LENC, N = BSZ * L, Lo = L / 2;
        aug_build<<<ew((size_t)N * 1536), 256, 0, stream>>>(X, Hh, BSZ, L, 512, 1536);
        gemm(Hh, 1536, in(46), 1536, 512, 1536, 512, in(47), nullptr, 0,
             R, nullptr, 512, N, 0);
        bn_elu_pool<<<ew((size_t)BSZ * Lo * 512), 256, 0, stream>>>(
            R, in(48), in(49), in(50), in(51), X, Xh, BSZ, L, Lo);
    }

    enc_layer(LENC / 2, 28, 36);
    layernorm_k<<<BSZ * (LENC / 2), 256, 0, stream>>>(X, in(52), in(53), X, Xh,
                                                      BSZ * (LENC / 2));

    // ---------------- decoder ----------------
    {
        int LE = LENC / 2;          // encoder output length (cross keys)
        int ND = BSZ * LDEC;        // 8 decoder tokens
        dec_emb_k<<<ew((size_t)ND * 512), 256, 0, stream>>>(in(2), in(7), in(3),
                                                            in(8), in(9), dX, dXh, BSZ, LDEC);
        // --- causal self attention (u == L_Q == 2 -> exact full attention) ---
        gemm(dXh, 512, in(54), 512, 512, 512, 512, in(55), nullptr, 0, dQ, nullptr, 512, ND, 0);
        gemm(dXh, 512, in(56), 512, 512, 512, 512, in(57), nullptr, 0, dK, nullptr, 512, ND, 0);
        gemm(dXh, 512, in(58), 512, 512, 512, 512, in(59), nullptr, 0, dV, nullptr, 512, ND, 0);
        dec_self_attn<<<ew(BSZ * NH * HD), 256, 0, stream>>>(dQ, dK, dV, dCTX, BSZ);
        cvt_half<<<ew((size_t)ND * 512), 256, 0, stream>>>(dCTX, dCTXh, (size_t)ND * 512);
        gemm(dCTXh, 512, in(60), 512, 512, 512, 512, in(61), dX, 512, dR, nullptr, 512, ND, 0);
        layernorm_k<<<ND, 256, 0, stream>>>(dR, in(74), in(75), dX, dXh, ND);
        // --- cross attention (u == L_Q == 2 -> exact full attention) ---
        gemm(dXh, 512, in(62), 512, 512, 512, 512, in(63), nullptr, 0, dQ, nullptr, 512, ND, 0);
        gemm(Xh, 512, in(64), 512, 512, 512, 512, in(65), nullptr, 0, Kc, nullptr, 512, BSZ * LE, 0);
        gemm(Xh, 512, in(66), 512, 512, 512, 512, in(67), nullptr, 0, Vc, nullptr, 512, BSZ * LE, 0);
        attn_update<<<BSZ * NH * LDEC, 256, 0, stream>>>(dQ, Kc, Vc, nullptr, dCTX,
                                                         LDEC, LE, LDEC);
        cvt_half<<<ew((size_t)ND * 512), 256, 0, stream>>>(dCTX, dCTXh, (size_t)ND * 512);
        gemm(dCTXh, 512, in(68), 512, 512, 512, 512, in(69), dX, 512, dR, nullptr, 512, ND, 0);
        layernorm_k<<<ND, 256, 0, stream>>>(dR, in(76), in(77), dX, dXh, ND);
        // --- FFN ---
        gemm(dXh, 512, in(70), 512, 2048, 512, 2048, in(71), nullptr, 0,
             nullptr, dHh, 2048, ND, 1);
        gemm(dHh, 2048, in(72), 2048, 512, 2048, 512, in(73), dX, 512,
             dR, nullptr, 512, ND, 0);
        layernorm_k<<<ND, 256, 0, stream>>>(dR, in(78), in(79), dX, dXh, ND);
        // --- final norm + projection ---
        layernorm_k<<<ND, 256, 0, stream>>>(dX, in(80), in(81), dX, dXh, ND);
        gemm(dXh, 512, in(82), 512, 118, 512, 128, in(83), nullptr, 0,
             Pbuf, nullptr, 118, ND, 0);
        out_copy<<<ew(BSZ * 118), 256, 0, stream>>>(Pbuf, (float*)d_out, BSZ);
    }
}
